// SS2D_37726992728138
// MI455X (gfx1250) — matrix operand order repaired and checked by reading, not compiled, not run on hardware
//
#include <hip/hip_runtime.h>
#include <hip/hip_bf16.h>

typedef __attribute__((ext_vector_type(16))) _Float16 v16h;
typedef __attribute__((ext_vector_type(2)))  _Float16 h2;
typedef __attribute__((ext_vector_type(8)))  float    v8f;

#define B_SZ 8
#define C_SZ 192
#define D_IN 384
#define NSTATE 16
#define DTR 12
#define LSEQ 1024
#define CDT 44   // DTR + 2*NSTATE

#define TILE_M 64
#define TILE_N 128
#define TILE_K 32

// cross-scan index map: xs[b,k,d,l] = xia[b,d,lmap(k,l)]; cross-merge writes to same index.
__device__ __forceinline__ int lmap(int k, int l) {
  switch (k) {
    default:
    case 0: return l;
    case 1: return ((l & 31) << 5) | (l >> 5);           // H = W = 32 transpose
    case 2: return (LSEQ - 1) - l;
    case 3: { int m = (LSEQ - 1) - l; return ((m & 31) << 5) | (m >> 5); }
  }
}

// Generic batched WMMA GEMM: C[z] (MxN) = A (MxK, row-major) * B[z] (KxN, row-major)
// mapped==1: z = b*4 + kdir; A offset kdir*aStride; B columns gathered via lmap(kdir, n).
//
// LDS tiles are FRAGMENT-MAJOR: each lane reads its whole WMMA operand as one
// contiguous 32-byte chunk (2x ds_load_b128). Tiles are double-buffered and the
// next tile's global loads are issued (into VGPRs) before the current tile's
// WMMAs so global latency hides under matrix math; staging guards are branchless
// (clamped row + 0/1 scale) so loads batch with a single wait.
//
// A fragment slot for element (m, kk):  lane = khalf*16 + (m&15), khalf = (kk>>3)&1,
//   kl = kk - 8*khalf (in {0..7,16..23}), h = 8*(kl>>4) + (kl&15)   [h even for even kk]
// B fragment slot for element (kr, n):  lane = ((kr>>3)&1)*16 + (n&15), h = (kr&7) + 8*(kr>>4)
__global__ __launch_bounds__(256) void k_gemm_wmma(
    const float* __restrict__ A, const float* __restrict__ Bm, float* __restrict__ Cm,
    int M, int N, int Kd, int mapped, long aStride, long bStride, long cStride)
{
  __shared__ __align__(32) _Float16 As[2][TILE_M * TILE_K];   // [buf][mtile(4)][lane(32)][h(16)]
  __shared__ __align__(32) _Float16 Bs[2][TILE_K * TILE_N];   // [buf][ntile(8)][lane(32)][h(16)]

  const int tid = threadIdx.x;
  const int z   = blockIdx.z;
  int kdir = 0;
  const float* Ap; const float* Bp; float* Cp;
  if (mapped) {
    int b = z >> 2; kdir = z & 3;
    Ap = A + (long)kdir * aStride;
    Bp = Bm + (long)b * bStride;
    Cp = Cm + (long)z * cStride;
  } else {
    Ap = A;
    Bp = Bm + (long)z * bStride;
    Cp = Cm + (long)z * cStride;
  }

  const int m0 = blockIdx.y * TILE_M;
  const int n0 = blockIdx.x * TILE_N;

  const int wave = tid >> 5, lane = tid & 31;
  const int mt    = wave & 3;          // wave's 16-row M tile within 64
  const int nhalf = wave >> 2;         // 0/1 -> 64-col half of 128

  // ---- per-thread staging coordinates (loop-invariant) ----
  // A: 1024 k-pairs, 4 per thread
  long aRow[4]; float aScl[4]; int aDst[4]; int aKk[4];
  #pragma unroll
  for (int j = 0; j < 4; ++j) {
    int i = tid + 256 * j;
    int m  = i >> 4;                   // 0..63
    int kk = (i & 15) << 1;            // even k in 0..30
    int row = m0 + m;
    aScl[j] = (row < M) ? 1.0f : 0.0f;
    aRow[j] = (row < M) ? row : (M - 1);    // clamped, always valid
    aKk[j]  = kk;
    int khalf = (kk >> 3) & 1;
    int kl = kk - 8 * khalf;           // {0..7,16..23}
    int h  = 8 * (kl >> 4) + (kl & 15);
    aDst[j] = (((m >> 4) << 5) + (khalf << 4) + (m & 15)) * 16 + h;
  }
  // B: 2048 k-row-pairs, 8 per thread
  int bKr[8]; int bCol[8]; int bDst[8];
  #pragma unroll
  for (int j = 0; j < 8; ++j) {
    int i = tid + 256 * j;
    int q  = i >> 7;                   // 0..15 -> rows 2q,2q+1
    int nc = i & 127;
    int kr = q << 1;
    int col = n0 + nc;
    bCol[j] = mapped ? lmap(kdir, col) : col;
    bKr[j]  = kr;
    bDst[j] = (((nc >> 4) << 5) + (((kr >> 3) & 1) << 4) + (nc & 15)) * 16 + ((kr & 7) + ((kr >> 4) << 3));
  }

  v8f acc[4];
  #pragma unroll
  for (int j = 0; j < 4; ++j)
    #pragma unroll
    for (int e = 0; e < 8; ++e) acc[j][e] = 0.0f;

  float aR[4][2], bR[8][2];

  // prologue: load tile k0=0 into registers (branchless, batched loads)
  #pragma unroll
  for (int j = 0; j < 4; ++j) {
    const float* p = &Ap[aRow[j] * (long)Kd + aKk[j]];
    aR[j][0] = p[0]; aR[j][1] = p[1];
  }
  #pragma unroll
  for (int j = 0; j < 8; ++j) {
    const float* p = &Bp[(long)bKr[j] * N + bCol[j]];
    bR[j][0] = p[0]; bR[j][1] = p[(size_t)N];
  }

  int buf = 0;
  for (int k0 = 0; k0 < Kd; k0 += TILE_K) {
    // ---- store staged registers -> LDS[buf] (f32 -> f16, 4-byte stores) ----
    #pragma unroll
    for (int j = 0; j < 4; ++j) {
      h2 pr; pr[0] = (_Float16)(aR[j][0] * aScl[j]);
             pr[1] = (_Float16)(aR[j][1] * aScl[j]);
      *(h2*)&As[buf][aDst[j]] = pr;
    }
    #pragma unroll
    for (int j = 0; j < 8; ++j) {
      h2 pr; pr[0] = (_Float16)bR[j][0];
             pr[1] = (_Float16)bR[j][1];
      *(h2*)&Bs[buf][bDst[j]] = pr;
    }
    __syncthreads();

    // ---- issue next tile's global loads; latency hides under WMMAs below ----
    if (k0 + TILE_K < Kd) {
      int kn = k0 + TILE_K;
      #pragma unroll
      for (int j = 0; j < 4; ++j) {
        const float* p = &Ap[aRow[j] * (long)Kd + (kn + aKk[j])];
        aR[j][0] = p[0]; aR[j][1] = p[1];
      }
      #pragma unroll
      for (int j = 0; j < 8; ++j) {
        const float* p = &Bp[(long)(kn + bKr[j]) * N + bCol[j]];
        bR[j][0] = p[0]; bR[j][1] = p[(size_t)N];
      }
    }

    // ---- fragments: one contiguous 32B read per lane per operand ----
    v16h afrag = *(const v16h*)&As[buf][((mt << 5) + lane) << 4];
    #pragma unroll
    for (int j = 0; j < 4; ++j) {
      int ntile = (nhalf << 2) + j;
      v16h bfrag = *(const v16h*)&Bs[buf][((ntile << 5) + lane) << 4];
      acc[j] = __builtin_amdgcn_wmma_f32_16x16x32_f16(
          false, afrag, false, bfrag, (short)0, acc[j], false, false);
    }
    buf ^= 1;
  }

  // store C: VGPR r -> row = m0 + mt*16 + r + 8*(lane>=16), col = n0 + ... + lane&15
  const int rbase = m0 + mt * 16 + 8 * (lane >> 4);
  #pragma unroll
  for (int j = 0; j < 4; ++j) {
    int gn = n0 + nhalf * 64 + 16 * j + (lane & 15);
    #pragma unroll
    for (int r = 0; r < 8; ++r) {
      int row = rbase + r;
      if (row < M) Cp[(long)row * N + gn] = acc[j][r];
    }
  }
}

// depthwise 3x3 conv (pad 1) + SiLU
__global__ __launch_bounds__(256) void k_conv_silu(
    const float* __restrict__ xi, const float* __restrict__ cw, float* __restrict__ out)
{
  long idx = (long)blockIdx.x * 256 + threadIdx.x;   // < B*D_IN*LSEQ
  int l = (int)(idx & (LSEQ - 1));
  long bd = idx >> 10;
  int d = (int)(bd % D_IN);
  int h = l >> 5, w = l & 31;
  const float* wp = cw + d * 9;
  const float* base = xi + bd * LSEQ;
  float acc = 0.0f;
  #pragma unroll
  for (int i = 0; i < 3; ++i) {
    int hh = h + i - 1;
    if ((unsigned)hh >= 32u) continue;
    #pragma unroll
    for (int j = 0; j < 3; ++j) {
      int ww = w + j - 1;
      if ((unsigned)ww >= 32u) continue;
      acc += wp[i * 3 + j] * base[hh * 32 + ww];
    }
  }
  out[idx] = acc / (1.0f + __expf(-acc));            // silu
}

__global__ void k_zero(float* __restrict__ p, long n)
{
  long i = (long)blockIdx.x * 256 + threadIdx.x;
  if (i < n) p[i] = 0.0f;
}

// selective scan: one block per (b,k); thread d owns 16-state recurrence in VGPRs.
// dt projection (rank 12) folded in; cross-merge fused via atomicAdd at mapped index.
__global__ __launch_bounds__(384) void k_scan(
    const float* __restrict__ xdbl, const float* __restrict__ xia,
    const float* __restrict__ Alog, const float* __restrict__ dtw,
    const float* __restrict__ dtb, const float* __restrict__ Dsv,
    float* __restrict__ yacc)
{
  __shared__ float xd[CDT * 64];                     // 64-step chunk of x_dbl rows
  const int bk = blockIdx.x;                         // b*4 + k
  const int b = bk >> 2, k = bk & 3;
  const int d = threadIdx.x;                         // 0..383
  const int gd = k * D_IN + d;

  float Ad[NSTATE], hs[NSTATE], wr[DTR];
  #pragma unroll
  for (int n = 0; n < NSTATE; ++n) {
    Ad[n] = -__expf(Alog[gd * NSTATE + n]);
    hs[n] = 0.0f;
  }
  #pragma unroll
  for (int r = 0; r < DTR; ++r) wr[r] = dtw[gd * DTR + r];
  const float bias = dtb[gd];
  const float Dv   = Dsv[gd];

  const float* xrow = xia + ((long)b * D_IN + d) * LSEQ;
  float*       yrow = yacc + ((long)b * D_IN + d) * LSEQ;
  const float* xdb  = xdbl + (long)bk * CDT * LSEQ;

  for (int l0 = 0; l0 < LSEQ; l0 += 64) {
    for (int i = d; i < CDT * 64; i += D_IN) {
      int c = i >> 6, j = i & 63;
      xd[i] = xdb[c * LSEQ + l0 + j];
    }
    __syncthreads();
    for (int j = 0; j < 64; ++j) {
      int l = l0 + j;
      float dt = bias;
      #pragma unroll
      for (int r = 0; r < DTR; ++r) dt += wr[r] * xd[r * 64 + j];
      float delta = (dt > 20.0f) ? dt : log1pf(__expf(dt));   // softplus
      int lm = lmap(k, l);
      float xs_v = xrow[lm];
      float u = delta * xs_v;
      float y = 0.0f;
      #pragma unroll
      for (int n = 0; n < NSTATE; ++n) {
        hs[n] = __expf(delta * Ad[n]) * hs[n] + u * xd[(DTR + n) * 64 + j];
        y += hs[n] * xd[(DTR + NSTATE + n) * 64 + j];
      }
      atomicAdd(&yrow[lm], y + Dv * xs_v);
    }
    __syncthreads();
  }
}

// LayerNorm over D_IN at each (b,l); wave32 per row, shfl_xor reduction.
__global__ __launch_bounds__(256) void k_ln(
    const float* __restrict__ y, const float* __restrict__ w, const float* __restrict__ bb,
    float* __restrict__ out)
{
  const int lane = threadIdx.x & 31;
  const int row  = blockIdx.x * 8 + (threadIdx.x >> 5);  // 0..8191 = b*1024 + l
  const int b = row >> 10, l = row & (LSEQ - 1);
  const long base = ((long)b * D_IN) * LSEQ + l;

  float v[12], s = 0.0f, ss = 0.0f;
  #pragma unroll
  for (int i = 0; i < 12; ++i) {
    int d = lane + 32 * i;
    v[i] = y[base + (long)d * LSEQ];
    s += v[i]; ss += v[i] * v[i];
  }
  #pragma unroll
  for (int off = 16; off > 0; off >>= 1) {
    s  += __shfl_xor(s,  off);
    ss += __shfl_xor(ss, off);
  }
  const float mean = s * (1.0f / D_IN);
  const float var  = ss * (1.0f / D_IN) - mean * mean;
  const float rstd = rsqrtf(var + 1e-5f);
  #pragma unroll
  for (int i = 0; i < 12; ++i) {
    int d = lane + 32 * i;
    out[base + (long)d * LSEQ] = (v[i] - mean) * rstd * w[d] + bb[d];
  }
}

extern "C" void kernel_launch(void* const* d_in, const int* in_sizes, int n_in,
                              void* d_out, int out_size, void* d_ws, size_t ws_size,
                              hipStream_t stream) {
  (void)in_sizes; (void)n_in; (void)out_size; (void)ws_size;
  const float* x     = (const float*)d_in[0];
  const float* w_in  = (const float*)d_in[1];
  const float* cw    = (const float*)d_in[2];
  const float* xpw   = (const float*)d_in[3];
  const float* dtw   = (const float*)d_in[4];
  const float* dtb   = (const float*)d_in[5];
  const float* Alog  = (const float*)d_in[6];
  const float* Dsv   = (const float*)d_in[7];
  const float* lnw   = (const float*)d_in[8];
  const float* lnb   = (const float*)d_in[9];
  const float* w_out = (const float*)d_in[10];
  float* out = (float*)d_out;

  const long nAct = (long)B_SZ * D_IN * LSEQ;        // 3,145,728
  float* xi   = (float*)d_ws;                        // in_proj output; reused as y-accumulator
  float* xia  = xi + nAct;                           // conv+silu output; reused as LN output
  float* xdbl = xia + nAct;                          // B*K*44*1024

  // 1. in_proj GEMM: (384x192) x (192x1024) per batch
  dim3 g1(LSEQ / TILE_N, D_IN / TILE_M, B_SZ);
  k_gemm_wmma<<<g1, 256, 0, stream>>>(w_in, x, xi, D_IN, LSEQ, C_SZ,
                                      0, 0, (long)C_SZ * LSEQ, (long)D_IN * LSEQ);
  // 2. depthwise conv 3x3 + SiLU
  k_conv_silu<<<(int)(nAct / 256), 256, 0, stream>>>(xi, cw, xia);
  // 3. zero y accumulator (reuse xi buffer)
  k_zero<<<(int)((nAct + 255) / 256), 256, 0, stream>>>(xi, nAct);
  // 4. x_dbl GEMM with fused cross-scan gather: (44x384) x (384x1024) per (b,k)
  dim3 g4(LSEQ / TILE_N, 1, B_SZ * 4);
  k_gemm_wmma<<<g4, 256, 0, stream>>>(xpw, xia, xdbl, CDT, LSEQ, D_IN,
                                      1, (long)CDT * D_IN, (long)D_IN * LSEQ, (long)CDT * LSEQ);
  // 5. selective scan with fused dt-projection, softplus, skip-D and cross-merge
  k_scan<<<B_SZ * 4, D_IN, 0, stream>>>(xdbl, xia, Alog, dtw, dtb, Dsv, xi);
  // 6. LayerNorm: y(xi) -> ynorm(xia)
  k_ln<<<(B_SZ * LSEQ) / 8, 256, 0, stream>>>(xi, lnw, lnb, xia);
  // 7. out_proj GEMM: (192x384) x (384x1024) per batch -> d_out
  dim3 g7(LSEQ / TILE_N, C_SZ / TILE_M, B_SZ);
  k_gemm_wmma<<<g7, 256, 0, stream>>>(w_out, xia, out, C_SZ, LSEQ, D_IN,
                                      0, 0, (long)D_IN * LSEQ, (long)C_SZ * LSEQ);
}
